// GIN_6219112644608
// MI455X (gfx1250) — matrix operand order repaired and checked by reading, not compiled, not run on hardware
//
#include <hip/hip_runtime.h>
#include <hip/hip_bf16.h>

#define N_NODES  50000
#define N_EDGES  800000
#define N_GRAPHS 512

typedef __attribute__((ext_vector_type(16))) __bf16 v16bf;
typedef __attribute__((ext_vector_type(8)))  float  v8f;

union ABFrag {
    v16bf v;
    uint4 q[2];
    unsigned short s[16];
};

__device__ __forceinline__ unsigned short f2bf(float f) {
    unsigned u = __float_as_uint(f);
    u = (u + 0x7FFFu + ((u >> 16) & 1u)) >> 16;   // round-to-nearest-even
    return (unsigned short)u;
}

// ---------------------------------------------------------------- utility
__global__ void k_copy_f32(const float* __restrict__ src, float* __restrict__ dst,
                           long long n) {
    long long i = (long long)blockIdx.x * blockDim.x + threadIdx.x;
    long long st = (long long)gridDim.x * blockDim.x;
    for (; i < n; i += st) dst[i] = src[i];
}

__global__ void k_zero_f32(float* __restrict__ dst, long long n) {
    long long i = (long long)blockIdx.x * blockDim.x + threadIdx.x;
    long long st = (long long)gridDim.x * blockDim.x;
    for (; i < n; i += st) dst[i] = 0.0f;
}

__global__ void k_f32_to_bf16(const float* __restrict__ src,
                              unsigned short* __restrict__ dst, long long n) {
    long long i = (long long)blockIdx.x * blockDim.x + threadIdx.x;
    long long st = (long long)gridDim.x * blockDim.x;
    for (; i < n; i += st) dst[i] = f2bf(src[i]);
}

// ------------------------------------------------- edge scatter (agg += x[src])
// one block per edge, blockDim.x == F ; all traffic stays in the 192MB L2
__global__ void k_edge_scatter(const float* __restrict__ x,
                               const int* __restrict__ ei,  // [2, N_EDGES]
                               float* __restrict__ agg, int F) {
    int e = blockIdx.x;
    int s = ei[e];
    int d = ei[N_EDGES + e];
    int f = threadIdx.x;
    atomicAdd(&agg[(size_t)d * F + f], x[(size_t)s * F + f]);
}

// ------------------------------------------------- pack W (K x N f32, row-major)
// into per-lane WMMA B-fragment order (bf16): tile t=(kt*(N/16)+nt), lane l holds
// col n = nt*16 + l%16, K = kt*32 + (l/16)*16 + e, e=0..15, contiguous 32B/lane.
__global__ void k_pack_w(const float* __restrict__ W, unsigned short* __restrict__ Wp,
                         int K, int N) {
    int t = blockIdx.x;           // (K/32)*(N/16) tiles
    int lane = threadIdx.x;       // 32 threads
    int ntiles = N >> 4;
    int kt = t / ntiles;
    int nt = t - kt * ntiles;
    int n = nt * 16 + (lane & 15);
    int kbase = kt * 32 + (lane >> 4) * 16;
    unsigned short* outp = Wp + ((size_t)t * 32 + lane) * 16;
#pragma unroll
    for (int e = 0; e < 16; ++e)
        outp[e] = f2bf(W[(size_t)(kbase + e) * N + n]);
}

// ------------------------------------------------- fused GIN MLP (both GEMMs)
// block = 256 threads = 8 waves; wave w owns n-tile w (16 of 128 output cols);
// block handles 16 node rows. hidden tile staged in LDS as bf16.
__global__ __launch_bounds__(256) void k_gin_mlp(
    const unsigned short* __restrict__ Abf, int ldA, int ktiles,
    const unsigned short* __restrict__ W1p, const float* __restrict__ b1,
    const unsigned short* __restrict__ W2p, const float* __restrict__ b2,
    float* __restrict__ out) {
    __shared__ unsigned short ldsH[16 * 128];   // 16 rows x 128 hidden (bf16)

    const int lane = threadIdx.x & 31;
    const int w    = threadIdx.x >> 5;   // 0..7 : n-tile
    const int half = lane >> 4;          // 0/1
    const int col  = lane & 15;
    const int m0   = blockIdx.x * 16;

    // -------- GEMM1: hidden = relu(A @ W1 + b1), K = ldA
    v8f c = {0.f, 0.f, 0.f, 0.f, 0.f, 0.f, 0.f, 0.f};
    const unsigned short* arow = Abf + (size_t)(m0 + col) * ldA;
    for (int kt = 0; kt < ktiles; ++kt) {
        ABFrag a, b;
        const unsigned short* ap = arow + kt * 32 + half * 16;
        a.q[0] = *(const uint4*)(ap);        // K = half*16 .. +7
        a.q[1] = *(const uint4*)(ap + 8);    // K = half*16 + 8 .. +15
        const uint4* wp = (const uint4*)(W1p + ((size_t)(kt * 8 + w) * 32 + lane) * 16);
        b.q[0] = wp[0];
        b.q[1] = wp[1];
        c = __builtin_amdgcn_wmma_f32_16x16x32_bf16(false, a.v, false, b.v,
                                                    (short)0, c, false, false);
    }
    const float bias1 = b1[w * 16 + col];
#pragma unroll
    for (int r = 0; r < 8; ++r) {
        float v = c[r] + bias1;
        v = v > 0.f ? v : 0.f;                        // inner ReLU
        ldsH[(half * 8 + r) * 128 + w * 16 + col] = f2bf(v);
    }
    __syncthreads();

    // -------- GEMM2: out = relu(hidden @ W2 + b2), K = 128
    v8f c2 = {0.f, 0.f, 0.f, 0.f, 0.f, 0.f, 0.f, 0.f};
    const unsigned short* hrow = ldsH + col * 128;
#pragma unroll
    for (int kt = 0; kt < 4; ++kt) {
        ABFrag a, b;
        const unsigned short* ap = hrow + kt * 32 + half * 16;
        a.q[0] = *(const uint4*)(ap);
        a.q[1] = *(const uint4*)(ap + 8);
        const uint4* wp = (const uint4*)(W2p + ((size_t)(kt * 8 + w) * 32 + lane) * 16);
        b.q[0] = wp[0];
        b.q[1] = wp[1];
        c2 = __builtin_amdgcn_wmma_f32_16x16x32_bf16(false, a.v, false, b.v,
                                                     (short)0, c2, false, false);
    }
    const float bias2 = b2[w * 16 + col];
#pragma unroll
    for (int r = 0; r < 8; ++r) {
        float v = c2[r] + bias2;
        v = v > 0.f ? v : 0.f;                        // outer ReLU (reference)
        out[(size_t)(m0 + half * 8 + r) * 128 + w * 16 + col] = v;
    }
}

// ------------------------------------------------- global_add_pool
__global__ void k_pool_scatter(const float* __restrict__ h,
                               const int* __restrict__ batch,
                               float* __restrict__ g) {
    int i = blockIdx.x;          // node
    int f = threadIdx.x;         // 128 feats
    atomicAdd(&g[(size_t)batch[i] * 128 + f], h[(size_t)i * 128 + f]);
}

// ------------------------------------------------- FC head + log_softmax
__global__ __launch_bounds__(64) void k_head(const float* __restrict__ g,
                                             const float* __restrict__ fc1W,
                                             const float* __restrict__ fc1b,
                                             const float* __restrict__ fc2W,
                                             const float* __restrict__ fc2b,
                                             float* __restrict__ out) {
    __shared__ float gg[128];
    __shared__ float hid[64];
    __shared__ float lg[2];
    int gi = blockIdx.x;
    int t = threadIdx.x;
    gg[t]      = g[(size_t)gi * 128 + t];
    gg[t + 64] = g[(size_t)gi * 128 + t + 64];
    __syncthreads();
    float acc = fc1b[t];
#pragma unroll 4
    for (int k = 0; k < 128; ++k) acc += gg[k] * fc1W[k * 64 + t];
    hid[t] = acc > 0.f ? acc : 0.f;
    __syncthreads();
    if (t < 2) {
        float l = fc2b[t];
        for (int j = 0; j < 64; ++j) l += hid[j] * fc2W[j * 2 + t];
        lg[t] = l;
    }
    __syncthreads();
    if (t == 0) {
        float m = fmaxf(lg[0], lg[1]);
        float lse = m + logf(expf(lg[0] - m) + expf(lg[1] - m));
        out[gi * 2 + 0] = lg[0] - lse;
        out[gi * 2 + 1] = lg[1] - lse;
    }
}

// =================================================================== launch
extern "C" void kernel_launch(void* const* d_in, const int* in_sizes, int n_in,
                              void* d_out, int out_size, void* d_ws, size_t ws_size,
                              hipStream_t stream) {
    const float* x       = (const float*)d_in[0];
    const int*   ei      = (const int*)d_in[1];
    const int*   batch   = (const int*)d_in[2];
    const float* cW1[3]  = {(const float*)d_in[3], (const float*)d_in[7],  (const float*)d_in[11]};
    const float* cb1[3]  = {(const float*)d_in[4], (const float*)d_in[8],  (const float*)d_in[12]};
    const float* cW2[3]  = {(const float*)d_in[5], (const float*)d_in[9],  (const float*)d_in[13]};
    const float* cb2[3]  = {(const float*)d_in[6], (const float*)d_in[10], (const float*)d_in[14]};
    const float* fc1W    = (const float*)d_in[15];
    const float* fc1b    = (const float*)d_in[16];
    const float* fc2W    = (const float*)d_in[17];
    const float* fc2b    = (const float*)d_in[18];
    float* out = (float*)d_out;

    // workspace carve-up (all 256B aligned)
    char* ws = (char*)d_ws;
    const size_t AGG_ELEMS = (size_t)N_NODES * 384;
    size_t off = 0;
    float*          agg  = (float*)(ws + off);          off += AGG_ELEMS * 4;            // 76.8 MB
    unsigned short* aggb = (unsigned short*)(ws + off); off += AGG_ELEMS * 2;            // 38.4 MB
    float*          h    = (float*)(ws + off);          off += (size_t)N_NODES * 128 * 4; // 25.6 MB
    float*          g    = (float*)(ws + off);          off += (size_t)N_GRAPHS * 128 * 4;
    unsigned short* Wp   = (unsigned short*)(ws + off);
    // packed weight sub-offsets (ushort units)
    unsigned short* W1p[3];
    unsigned short* W2p[3];
    W1p[0] = Wp;                 // 384x128 -> 49152
    W2p[0] = Wp + 49152;         // 128x128 -> 16384
    W1p[1] = Wp + 65536;
    W2p[1] = Wp + 81920;
    W1p[2] = Wp + 98304;
    W2p[2] = Wp + 114688;

    // ---- pack weights into WMMA B-fragment order (bf16)
    k_pack_w<<<(384 / 32) * (128 / 16), 32, 0, stream>>>(cW1[0], W1p[0], 384, 128);
    k_pack_w<<<(128 / 32) * (128 / 16), 32, 0, stream>>>(cW2[0], W2p[0], 128, 128);
    for (int l = 1; l < 3; ++l) {
        k_pack_w<<<(128 / 32) * (128 / 16), 32, 0, stream>>>(cW1[l], W1p[l], 128, 128);
        k_pack_w<<<(128 / 32) * (128 / 16), 32, 0, stream>>>(cW2[l], W2p[l], 128, 128);
    }

    const int CP_BLOCKS = 2048, CP_THREADS = 256;
    const int MLP_BLOCKS = N_NODES / 16;   // 50000/16 = 3125 exact

    // ---- layer 1 (F_in = 384)
    {
        long long n = (long long)N_NODES * 384;
        k_copy_f32<<<CP_BLOCKS, CP_THREADS, 0, stream>>>(x, agg, n);       // (1+eps)*x, eps=0
        k_edge_scatter<<<N_EDGES, 384, 0, stream>>>(x, ei, agg, 384);
        k_f32_to_bf16<<<CP_BLOCKS, CP_THREADS, 0, stream>>>(agg, aggb, n);
        k_gin_mlp<<<MLP_BLOCKS, 256, 0, stream>>>(aggb, 384, 384 / 32,
                                                  W1p[0], cb1[0], W2p[0], cb2[0], h);
    }
    // ---- layers 2,3 (F_in = 128)
    for (int l = 1; l < 3; ++l) {
        long long n = (long long)N_NODES * 128;
        k_copy_f32<<<CP_BLOCKS, CP_THREADS, 0, stream>>>(h, agg, n);
        k_edge_scatter<<<N_EDGES, 128, 0, stream>>>(h, ei, agg, 128);
        k_f32_to_bf16<<<CP_BLOCKS, CP_THREADS, 0, stream>>>(agg, aggb, n);
        k_gin_mlp<<<MLP_BLOCKS, 256, 0, stream>>>(aggb, 128, 128 / 32,
                                                  W1p[l], cb1[l], W2p[l], cb2[l], h);
    }

    // ---- pooling + head
    k_zero_f32<<<64, 256, 0, stream>>>(g, (long long)N_GRAPHS * 128);
    k_pool_scatter<<<N_NODES, 128, 0, stream>>>(h, batch, g);
    k_head<<<N_GRAPHS, 64, 0, stream>>>(g, fc1W, fc1b, fc2W, fc2b, out);

    (void)in_sizes; (void)n_in; (void)out_size; (void)ws_size;
}
